// Peony_60498909331609
// MI455X (gfx1250) — matrix operand order repaired and checked by reading, not compiled, not run on hardware
//
#include <hip/hip_runtime.h>
#include <stdint.h>

// ---------------- problem constants ----------------
#define Bm     16
#define Tm     2048
#define BT     32768            // Bm*Tm
#define NMELS  80
#define NMELSP 96               // padded to multiple of 32 for WMMA K
#define DM     512
#define DB     64
#define NCODE  1024
#define NQ     8
#define KW     5
#define NL     4
#define NEGBIG (-1.0e9f)

// ---------------- vector types ----------------
typedef __bf16          v16bf  __attribute__((ext_vector_type(16)));
typedef float           v8f    __attribute__((ext_vector_type(8)));
typedef unsigned short  u16x16 __attribute__((ext_vector_type(16)));
typedef unsigned short  u16x8  __attribute__((ext_vector_type(8)));
typedef int             i32x4  __attribute__((ext_vector_type(4)));

// ---------------- small device helpers ----------------
static __device__ __forceinline__ unsigned short f2bf(float f) {
    unsigned int x = __float_as_uint(f);
    unsigned int r = x + 0x7FFFu + ((x >> 16) & 1u);   // round-to-nearest-even
    return (unsigned short)(r >> 16);
}

static __device__ __forceinline__ float wred_sum(float v) {
    #pragma unroll
    for (int o = 16; o > 0; o >>= 1) v += __shfl_xor(v, o, 32);
    return v;
}
static __device__ __forceinline__ float wred_max(float v) {
    #pragma unroll
    for (int o = 16; o > 0; o >>= 1) v = fmaxf(v, __shfl_xor(v, o, 32));
    return v;
}

// A fragment (16x32 bf16, MxK), branch-free. Lane (h = lane>>4, m = lane&15).
// Register pairs hold elems [k0+8h .. +7] then [k0+16+8h .. +7] (CDNA5 ISA 7.12.2).
static __device__ __forceinline__ u16x16 load_a16(const unsigned short* rowptr, int k0, int h) {
    const u16x8* p = (const u16x8*)(rowptr + k0 + 8 * h);
    u16x8 lo = p[0];          // k0+8h   .. +7
    u16x8 hi = p[2];          // k0+16+8h.. +7
    u16x16 r;
    #pragma unroll
    for (int j = 0; j < 8; j++) { r[j] = lo[j]; r[8 + j] = hi[j]; }
    return r;
}

// B fragment (32x16 bf16, KxN), row-major Bw[n][k] (K contiguous), global or LDS pointer.
static __device__ __forceinline__ u16x16 load_b16(const unsigned short* rowptr, int k0, int h) {
    const u16x8* p = (const u16x8*)(rowptr + k0 + 8 * h);
    u16x8 lo = p[0];
    u16x8 hi = p[2];
    u16x16 r;
    #pragma unroll
    for (int j = 0; j < 8; j++) { r[j] = lo[j]; r[8 + j] = hi[j]; }
    return r;
}

static __device__ __forceinline__ v8f wmma_bf16(u16x16 a, u16x16 b, v8f c) {
    return __builtin_amdgcn_wmma_f32_16x16x32_bf16(
        false, __builtin_bit_cast(v16bf, a),
        false, __builtin_bit_cast(v16bf, b),
        (short)0, c, false, false);
}

// Async global->LDS 16B copy (gfx1250); falls back to load+ds_store if unavailable.
// Builtin signature (from toolchain diagnostic): (int4 AS(1)*, int4 AS(3)*, imm offset, imm cpol).
static __device__ __forceinline__ void cp_async16(const unsigned short* gsrc, unsigned short* lds) {
#if __has_builtin(__builtin_amdgcn_global_load_async_to_lds_b128)
    __builtin_amdgcn_global_load_async_to_lds_b128(
        (__attribute__((address_space(1))) i32x4*)(__attribute__((address_space(1))) void*)(gsrc),
        (__attribute__((address_space(3))) i32x4*)(__attribute__((address_space(3))) void*)(lds),
        0, 0);
#else
    *(u16x8*)lds = *(const u16x8*)gsrc;
#endif
}
static __device__ __forceinline__ void cp_async_wait() {
#if __has_builtin(__builtin_amdgcn_s_wait_asynccnt)
    __builtin_amdgcn_s_wait_asynccnt(0);
#endif
}

// ---------------- elementwise / init kernels ----------------
__global__ void zero_k(float* p, int n) {
    int i = blockIdx.x * blockDim.x + threadIdx.x;
    if (i < n) p[i] = 0.0f;
}

// f32 -> bf16 with K padding: d is [rows, kdst], s is [rows, ksrc]; cols >= ksrc get 0.
__global__ void cvt_pad_k(const float* __restrict__ s, unsigned short* __restrict__ d,
                          int rows, int ksrc, int kdst) {
    int i = blockIdx.x * blockDim.x + threadIdx.x;
    int n = rows * kdst;
    if (i < n) {
        int r = i / kdst;
        int c = i - r * kdst;
        d[i] = (c < ksrc) ? f2bf(s[(size_t)r * ksrc + c]) : (unsigned short)0;
    }
}

// ---------------- generic bf16 WMMA GEMM: C[M,N] = A[M,K] * Bw[N,K]^T + bias ----------------
// Kdim must be a multiple of 32; A rows 16B aligned, Bw rows 32B aligned.
__global__ void gemm_bf16_k(const unsigned short* __restrict__ A, int lda,
                            const unsigned short* __restrict__ Bw, int ldb,
                            const float* __restrict__ bias,
                            float* __restrict__ Cf, unsigned short* __restrict__ Cbf,
                            int M, int N, int Kdim) {
    int wave = threadIdx.x >> 5;
    int lane = threadIdx.x & 31;
    int h  = lane >> 4;
    int ln = lane & 15;
    int n0 = blockIdx.x * 16;
    int m0 = (blockIdx.y * 8 + wave) * 16;

    const unsigned short* arow = A  + (size_t)(m0 + ln) * lda;
    const unsigned short* brow = Bw + (size_t)(n0 + ln) * ldb;

    v8f acc = {0.f, 0.f, 0.f, 0.f, 0.f, 0.f, 0.f, 0.f};
    #pragma unroll 4
    for (int k0 = 0; k0 < Kdim; k0 += 32) {
        u16x16 a = load_a16(arow, k0, h);
        u16x16 b = load_b16(brow, k0, h);
        acc = wmma_bf16(a, b, acc);
    }

    float bv = bias ? bias[n0 + ln] : 0.0f;
    #pragma unroll
    for (int r = 0; r < 8; r++) {
        int mm = m0 + r + 8 * h;
        float v = acc[r] + bv;
        size_t o = (size_t)mm * N + (n0 + ln);
        if (Cf)  Cf[o]  = v;
        if (Cbf) Cbf[o] = f2bf(v);
    }
}

// ---------------- weight-norm + transpose: wT[dk][co][ci] = (g/||v||)*v[co][ci][dk] ----------------
__global__ void wnorm_k(const float* __restrict__ v, const float* __restrict__ g,
                        unsigned short* __restrict__ wT) {
    int wave = threadIdx.x >> 5;
    int lane = threadIdx.x & 31;
    int co = blockIdx.x * 8 + wave;
    const float* vr = v + (size_t)co * DM * KW;
    float ss = 0.f;
    for (int j = lane; j < DM * KW; j += 32) { float x = vr[j]; ss += x * x; }
    ss = wred_sum(ss);
    float sc = g[co] / sqrtf(fmaxf(ss, 1e-24f));
    for (int ci = lane; ci < DM; ci += 32) {
        #pragma unroll
        for (int dk = 0; dk < KW; dk++)
            wT[((size_t)dk * DM + co) * DM + ci] = f2bf(sc * vr[ci * KW + dk]);
    }
}

// ---------------- implicit-GEMM conv1d (k=5, same pad) + bias + relu + BN stats ----------------
// Weight tile per tap staged into LDS via async copy (shared by all 8 waves of the block);
// activation padding handled via `zrow` (512 zeroed bf16) -> branch-free A fragment loads.
__global__ void __launch_bounds__(256)
conv_wmma_k(const unsigned short* __restrict__ X,   // [B*T, 512] bf16
            const unsigned short* __restrict__ Wt,  // [5, 512(co), 512(ci)] bf16
            const unsigned short* __restrict__ zrow,
            const float* __restrict__ bias,
            float* __restrict__ Y,                  // [B*T, 512] f32 (post relu)
            float* __restrict__ csum, float* __restrict__ csumsq) {
    __shared__ __align__(32) unsigned short wlds[16 * DM];   // one tap: 16 co x 512 ci (16 KB)

    int tid  = threadIdx.x;
    int wave = tid >> 5;
    int lane = tid & 31;
    int h  = lane >> 4;
    int ln = lane & 15;
    int co0   = blockIdx.x * 16;
    int mtile = blockIdx.y * 8 + wave;
    int b  = mtile >> 7;            // T/16 = 128 tiles per batch
    int t0 = (mtile & 127) << 4;
    int tA = t0 + ln;               // A-row time index for this lane

    // precompute (clamped) A row pointers for all 5 taps
    const unsigned short* arows[KW];
    #pragma unroll
    for (int dk = 0; dk < KW; dk++) {
        int tt = tA + dk - 2;
        bool ok = (tt >= 0) && (tt < Tm);
        arows[dk] = ok ? (X + ((size_t)b * Tm + tt) * DM) : zrow;   // pointer select, no branch
    }

    v8f acc = {0.f, 0.f, 0.f, 0.f, 0.f, 0.f, 0.f, 0.f};
    for (int dk = 0; dk < KW; dk++) {
        // ---- stage this tap's 16x512 bf16 weight tile into LDS (async) ----
        if (dk) __syncthreads();                      // previous tile fully consumed
        const unsigned short* wsrc = Wt + (size_t)(dk * DM + co0) * DM;
        #pragma unroll
        for (int c = 0; c < (16 * DM) / (8 * 256); c++) {       // 4 x 16B chunks per thread
            int e = (c * 256 + tid) * 8;
            cp_async16(wsrc + e, &wlds[e]);
        }
        cp_async_wait();
        __syncthreads();

        const unsigned short* arow = arows[dk];
        const unsigned short* brow = &wlds[ln * DM];
        if (dk + 1 < KW)
            __builtin_prefetch(arows[dk + 1], 0, 3);   // global_prefetch of next tap's row
        #pragma unroll 4
        for (int k0 = 0; k0 < DM; k0 += 32) {
            u16x16 a  = load_a16(arow, k0, h);        // global_load_b128 x2
            u16x16 bb = load_b16(brow, k0, h);        // ds_load_b128 x2
            acc = wmma_bf16(a, bb, acc);
        }
    }

    float bv = bias[co0 + ln];
    float s = 0.f, s2 = 0.f;
    #pragma unroll
    for (int r = 0; r < 8; r++) {
        int t = t0 + r + 8 * h;
        float v = fmaxf(acc[r] + bv, 0.0f);
        Y[((size_t)b * Tm + t) * DM + co0 + ln] = v;
        s += v; s2 += v * v;
    }
    s  += __shfl_xor(s, 16, 32);
    s2 += __shfl_xor(s2, 16, 32);
    if (h == 0) {
        atomicAdd(&csum[co0 + ln], s);
        atomicAdd(&csumsq[co0 + ln], s2);
    }
}

__global__ void bn_finalize_k(float* csum, float* csumsq,
                              const float* __restrict__ gamma, const float* __restrict__ beta,
                              float* __restrict__ scale, float* __restrict__ shift,
                              float inv_count) {
    int c = blockIdx.x * blockDim.x + threadIdx.x;
    if (c < DM) {
        float m   = csum[c] * inv_count;
        float var = csumsq[c] * inv_count - m * m;
        float sc  = gamma[c] / sqrtf(var + 1e-5f);
        scale[c] = sc;
        shift[c] = beta[c] - m * sc;
        csum[c] = 0.f; csumsq[c] = 0.f;       // ready for next layer
    }
}

__global__ void bn_apply_k(const float* __restrict__ Y,
                           const float* __restrict__ scale, const float* __restrict__ shift,
                           unsigned short* __restrict__ Xbf, int n) {
    int i = blockIdx.x * blockDim.x + threadIdx.x;
    if (i < n) {
        int c = i & (DM - 1);
        Xbf[i] = f2bf(Y[i] * scale[c] + shift[c]);
    }
}

// ---------------- mask + l2norm of z (wave per (b,t) row of 64) ----------------
__global__ void mask_norm_k(float* __restrict__ z, const unsigned char* __restrict__ mask,
                            unsigned short* __restrict__ zbf,
                            float* __restrict__ znf, unsigned short* __restrict__ znbf,
                            float* __restrict__ zsq) {
    int wave = threadIdx.x >> 5;
    int lane = threadIdx.x & 31;
    int r = blockIdx.x * 8 + wave;
    float mf = mask[r] ? 1.0f : 0.0f;
    size_t base = (size_t)r * DB;
    float v0 = z[base + lane] * mf;
    float v1 = z[base + 32 + lane] * mf;
    float ss = wred_sum(v0 * v0 + v1 * v1);
    float inv = 1.0f / fmaxf(sqrtf(ss), 1e-12f);
    float n0 = v0 * inv, n1 = v1 * inv;
    z[base + lane] = v0;          z[base + 32 + lane] = v1;
    zbf[base + lane] = f2bf(v0);  zbf[base + 32 + lane] = f2bf(v1);
    znf[base + lane] = n0;        znf[base + 32 + lane] = n1;
    znbf[base + lane] = f2bf(n0); znbf[base + 32 + lane] = f2bf(n1);
    float q = wred_sum(n0 * n0 + n1 * n1);
    if (lane == 0) zsq[r] = q;
}

// ---------------- codebook l2norm (wave per (q,code)) ----------------
__global__ void cb_prep_k(const float* __restrict__ cb, float* __restrict__ cbn,
                          unsigned short* __restrict__ cbnbf, float* __restrict__ cbsq) {
    int wave = threadIdx.x >> 5;
    int lane = threadIdx.x & 31;
    int row = blockIdx.x * 8 + wave;          // NQ*NCODE rows
    size_t base = (size_t)row * DB;
    float v0 = cb[base + lane], v1 = cb[base + 32 + lane];
    float ss = wred_sum(v0 * v0 + v1 * v1);
    float inv = 1.0f / fmaxf(sqrtf(ss), 1e-12f);
    v0 *= inv; v1 *= inv;
    cbn[base + lane] = v0;         cbn[base + 32 + lane] = v1;
    cbnbf[base + lane] = f2bf(v0); cbnbf[base + 32 + lane] = f2bf(v1);
    float q = wred_sum(v0 * v0 + v1 * v1);
    if (lane == 0) cbsq[row] = q;
}

__global__ void msum_k(const unsigned char* __restrict__ mask, float* __restrict__ msum, int n) {
    float s = 0.f;
    for (int i = blockIdx.x * blockDim.x + threadIdx.x; i < n; i += gridDim.x * blockDim.x)
        s += mask[i] ? 1.0f : 0.0f;
    s = wred_sum(s);
    if ((threadIdx.x & 31) == 0) atomicAdd(msum, s);
}

// ---------------- VQ distance GEMM (WMMA): dist = z_sq - 2*zn@cb^T + cbsq, masked ----------------
__global__ void vq_dist_k(const unsigned short* __restrict__ znbf,  // [BT,64] bf16
                          const unsigned short* __restrict__ cbbf,  // [1024,64] bf16 (this q)
                          const float* __restrict__ z_sq, const float* __restrict__ cbsq,
                          const unsigned char* __restrict__ mask,
                          float* __restrict__ dist, int row0) {
    int wave = threadIdx.x >> 5;
    int lane = threadIdx.x & 31;
    int h  = lane >> 4;
    int ln = lane & 15;
    int n0  = blockIdx.x * 16;
    int lm0 = (blockIdx.y * 8 + wave) * 16;

    const unsigned short* arow = znbf + (size_t)(row0 + lm0 + ln) * DB;
    const unsigned short* brow = cbbf + (size_t)(n0 + ln) * DB;

    v8f acc = {0.f, 0.f, 0.f, 0.f, 0.f, 0.f, 0.f, 0.f};
    #pragma unroll
    for (int k0 = 0; k0 < DB; k0 += 32) {
        u16x16 a = load_a16(arow, k0, h);
        u16x16 b = load_b16(brow, k0, h);
        acc = wmma_bf16(a, b, acc);
    }

    float cq = cbsq[n0 + ln];
    #pragma unroll
    for (int r = 0; r < 8; r++) {
        int lmm = lm0 + r + 8 * h;
        int g = row0 + lmm;
        float d = mask[g] ? (z_sq[g] - 2.0f * acc[r] + cq) : NEGBIG;
        dist[(size_t)lmm * NCODE + (n0 + ln)] = d;
    }
}

// ---------------- VQ loss: softmax -> log_softmax(p) -> CE + commit (wave per row) ----------------
__global__ void vq_loss_k(const float* __restrict__ dist, const float* __restrict__ zn,
                          const float* __restrict__ cbnq, const int* __restrict__ iidx,
                          int q, const unsigned char* __restrict__ mask, int row0,
                          float* __restrict__ acc_ce, float* __restrict__ acc_commit) {
    int wave = threadIdx.x >> 5;
    int lane = threadIdx.x & 31;
    int lr = blockIdx.x * 8 + wave;   // local row in this chunk
    int r  = row0 + lr;               // global (b,t) row
    const float* drow = dist + (size_t)lr * NCODE;

    float mv = -3.4e38f;
    for (int c = lane; c < NCODE; c += 32) mv = fmaxf(mv, -drow[c]);
    mv = wred_max(mv);

    float s1 = 0.f;
    for (int c = lane; c < NCODE; c += 32) s1 += __expf(-drow[c] - mv);
    s1 = wred_sum(s1);
    float inv_s1 = 1.0f / s1;
    float pmax = inv_s1;              // max of p == exp(0)/s1

    float s2 = 0.f;
    for (int c = lane; c < NCODE; c += 32) {
        float p = __expf(-drow[c] - mv) * inv_s1;
        s2 += __expf(p - pmax);
    }
    s2 = wred_sum(s2);
    float lse = pmax + __logf(s2);

    int idx = iidx[(size_t)r * NQ + q];
    float pidx = __expf(-drow[idx] - mv) * inv_s1;
    float ce = lse - pidx;            // -(p[idx] - logsumexp(p))

    // commit = mean_d (zn - l2norm(cb[idx]))^2
    const float* code = cbnq + (size_t)idx * DB;
    float c0 = code[lane], c1 = code[32 + lane];
    float nn = wred_sum(c0 * c0 + c1 * c1);
    float inv = 1.0f / fmaxf(sqrtf(nn), 1e-12f);
    c0 *= inv; c1 *= inv;
    float z0 = zn[(size_t)r * DB + lane], z1 = zn[(size_t)r * DB + 32 + lane];
    float d0 = z0 - c0, d1 = z1 - c1;
    float cm = wred_sum(d0 * d0 + d1 * d1) * (1.0f / 64.0f);

    float mf = mask[r] ? 1.0f : 0.0f;
    if (lane == 0) {
        atomicAdd(acc_ce, ce * mf);
        atomicAdd(acc_commit, cm * mf);
    }
}

__global__ void finalize_k(const float* __restrict__ msum, const float* __restrict__ acc_ce,
                           const float* __restrict__ acc_cm, float* __restrict__ out3) {
    if (threadIdx.x == 0 && blockIdx.x == 0) {
        float ms = *msum;
        out3[0] = *acc_cm / ms;   // commitment_loss
        out3[1] = *acc_cm / ms;   // codebook_loss (identical in forward)
        out3[2] = *acc_ce / ms;   // label_loss
    }
}

// ---------------- host side ----------------
extern "C" void kernel_launch(void* const* d_in, const int* in_sizes, int n_in,
                              void* d_out, int out_size, void* d_ws, size_t ws_size,
                              hipStream_t stream) {
    (void)in_sizes; (void)n_in; (void)out_size; (void)ws_size;

    const float*         mel   = (const float*)d_in[0];
    const int*           iidx  = (const int*)d_in[1];
    const unsigned char* mask  = (const unsigned char*)d_in[2];
    auto P = [&](int j) { return (const float*)d_in[j]; };
    // enc_params base=3, dec_params base=27 (alphabetical flatten:
    // layers[l] = {b, beta, g, gamma, v}, then post_b, post_w, pre_b, pre_w)
    const float* codebooks = P(51);

    // ---- workspace carve-out ----
    char* wsb = (char*)d_ws;
    size_t off = 0;
    auto carve = [&](size_t bytes) { void* p = wsb + off; off += (bytes + 255) & ~(size_t)255; return p; };
    float*          act_f   = (float*)carve((size_t)BT * DM * 4);      // conv out; reused as dist chunk
    unsigned short* act_bf  = (unsigned short*)carve((size_t)BT * DM * 2);
    unsigned short* mel_bf  = (unsigned short*)carve((size_t)BT * NMELSP * 2);   // K padded 80->96
    float*          z_f     = (float*)carve((size_t)BT * DB * 4);
    unsigned short* z_bf    = (unsigned short*)carve((size_t)BT * DB * 2);
    float*          zn_f    = (float*)carve((size_t)BT * DB * 4);
    unsigned short* zn_bf   = (unsigned short*)carve((size_t)BT * DB * 2);
    float*          z_sq    = (float*)carve((size_t)BT * 4);
    float*          cbn_f   = (float*)carve((size_t)NQ * NCODE * DB * 4);
    unsigned short* cbn_bf  = (unsigned short*)carve((size_t)NQ * NCODE * DB * 2);
    float*          cbsq    = (float*)carve((size_t)NQ * NCODE * 4);
    unsigned short* wT      = (unsigned short*)carve((size_t)KW * DM * DM * 2);
    unsigned short* wpre_bf = (unsigned short*)carve((size_t)DM * NMELSP * 2);
    unsigned short* wpost_bf= (unsigned short*)carve((size_t)NMELS * DM * 2);
    unsigned short* zrow    = (unsigned short*)carve((size_t)DM * 2);  // zero guard row
    float*          bnscale = (float*)carve(DM * 4);
    float*          bnshift = (float*)carve(DM * 4);
    float*          accs    = (float*)carve((2 * DM + 3) * 4);
    float* csum = accs, *csumsq = accs + DM;
    float* msum = accs + 2 * DM, *acc_ce = accs + 2 * DM + 1, *acc_cm = accs + 2 * DM + 2;

    dim3 blk(256);
    auto cvt_pad = [&](const float* s, unsigned short* d, int rows, int ksrc, int kdst) {
        int n = rows * kdst;
        cvt_pad_k<<<dim3((n + 255) / 256), blk, 0, stream>>>(s, d, rows, ksrc, kdst);
    };

    // zero accumulators (csum, csumsq, msum, acc_ce, acc_cm) and the guard row (512 bf16 = 256 f32)
    zero_k<<<dim3((2 * DM + 3 + 255) / 256), blk, 0, stream>>>(accs, 2 * DM + 3);
    zero_k<<<dim3(1), blk, 0, stream>>>((float*)zrow, DM / 2);

    // ---- one conv_stack ----
    auto run_stack = [&](const unsigned short* in_bf, int kin_src, int kin_pad, int pbase,
                         int nout, float* out_f) {
        const float* pre_w  = P(pbase + 23);
        const float* pre_b  = P(pbase + 22);
        const float* post_w = P(pbase + 21);
        const float* post_b = P(pbase + 20);
        cvt_pad(pre_w, wpre_bf, DM, kin_src, kin_pad);
        gemm_bf16_k<<<dim3(DM / 16, BT / 128), blk, 0, stream>>>(
            in_bf, kin_pad, wpre_bf, kin_pad, pre_b, nullptr, act_bf, BT, DM, kin_pad);
        for (int l = 0; l < NL; l++) {
            const float* lb    = P(pbase + 5 * l + 0);
            const float* lbeta = P(pbase + 5 * l + 1);
            const float* lg    = P(pbase + 5 * l + 2);
            const float* lgam  = P(pbase + 5 * l + 3);
            const float* lv    = P(pbase + 5 * l + 4);
            wnorm_k<<<dim3(DM / 8), blk, 0, stream>>>(lv, lg, wT);
            conv_wmma_k<<<dim3(DM / 16, BT / 128), blk, 0, stream>>>(
                act_bf, wT, zrow, lb, act_f, csum, csumsq);
            bn_finalize_k<<<dim3(2), blk, 0, stream>>>(
                csum, csumsq, lgam, lbeta, bnscale, bnshift, 1.0f / (float)BT);
            bn_apply_k<<<dim3((BT * DM) / 256), blk, 0, stream>>>(
                act_f, bnscale, bnshift, act_bf, BT * DM);
        }
        cvt_pad(post_w, wpost_bf, nout, DM, DM);
        gemm_bf16_k<<<dim3((nout + 15) / 16, BT / 128), blk, 0, stream>>>(
            act_bf, DM, wpost_bf, DM, post_b, out_f, nullptr, BT, nout, DM);
    };

    // ================= encoder =================
    cvt_pad(mel, mel_bf, BT, NMELS, NMELSP);
    run_stack(mel_bf, NMELS, NMELSP, /*enc base*/3, DB, z_f);

    // mask + l2norm
    mask_norm_k<<<dim3(BT / 8), blk, 0, stream>>>(z_f, mask, z_bf, zn_f, zn_bf, z_sq);

    // codebooks + mask sum
    cb_prep_k<<<dim3((NQ * NCODE) / 8), blk, 0, stream>>>(codebooks, cbn_f, cbn_bf, cbsq);
    msum_k<<<dim3(64), blk, 0, stream>>>(mask, msum, BT);

    // ================= VQ losses (reuse act_f as dist scratch in 2 row-chunks) =================
    const int CHUNK = BT / 2;   // 16384 rows -> 64 MB dist chunk
    for (int q = 0; q < NQ; q++) {
        const unsigned short* cbbf_q = cbn_bf + (size_t)q * NCODE * DB;
        const float*          cbf_q  = cbn_f  + (size_t)q * NCODE * DB;
        const float*          cbsq_q = cbsq + (size_t)q * NCODE;
        for (int half = 0; half < 2; half++) {
            int row0 = half * CHUNK;
            vq_dist_k<<<dim3(NCODE / 16, CHUNK / 128), blk, 0, stream>>>(
                zn_bf, cbbf_q, z_sq, cbsq_q, mask, act_f, row0);
            vq_loss_k<<<dim3(CHUNK / 8), blk, 0, stream>>>(
                act_f, zn_f, cbf_q, iidx, q, mask, row0, acc_ce, acc_cm);
        }
    }

    // ================= decoder (input is masked z) =================
    run_stack(z_bf, DB, DB, /*dec base*/27, NMELS, (float*)d_out);

    // ================= scalar outputs =================
    finalize_k<<<dim3(1), dim3(32), 0, stream>>>(msum, acc_ce, acc_cm,
                                                 (float*)d_out + (size_t)BT * NMELS);
}
